// Fakenews_GNN_16303695856042
// MI455X (gfx1250) — compile-verified, matrix operand order checked by reading, not run on hardware
//
#include <hip/hip_runtime.h>
#include <hip/hip_bf16.h>

typedef __attribute__((ext_vector_type(16))) _Float16 v16h;
typedef __attribute__((ext_vector_type(8)))  float    v8f;

#define HDIM 64
#define MAX_KTILES 8   // K <= 256

// ---------------- degree / norm kernels ----------------

__global__ void deg_init_kernel(float* deg, int n) {
    int i = blockIdx.x * blockDim.x + threadIdx.x;
    if (i < n) deg[i] = 1.0f;   // self-loop contributes 1 to every node
}

__global__ void deg_accum_kernel(const int* __restrict__ ei, float* deg, int e) {
    int i = blockIdx.x * blockDim.x + threadIdx.x;
    if (i < e) atomicAdd(&deg[ei[(long)e + i]], 1.0f);   // dst row
}

__global__ void deg_rsqrt_kernel(float* deg, int n) {
    int i = blockIdx.x * blockDim.x + threadIdx.x;
    if (i < n) deg[i] = rsqrtf(deg[i]);   // deg >= 1 always (self loops)
}

// ---------------- B prepack: f32 [K x 64] -> fragment-ordered f16 ----------------
// Fragment (ktile, ntile, lane) holds 16 f16 contiguous (32 B): the exact per-lane
// WMMA B operand. lane<16 -> K = {0..7, 16..23}+k0, lane>=16 -> K = {8..15, 24..31}+k0,
// col = ntile*16 + (lane & 15).

__global__ void pack_b_kernel(const float* __restrict__ B, _Float16* __restrict__ Bp,
                              int K) {
    int t = blockIdx.x * blockDim.x + threadIdx.x;
    int total = (K >> 5) * 4 * 32;            // (ktiles)*(ntiles)*(lanes)
    if (t >= total) return;
    int lane  = t & 31;
    int ntile = (t >> 5) & 3;
    int ktile = t >> 7;
    int k0    = ktile << 5;
    int col   = (ntile << 4) + (lane & 15);
    int halfsel = (lane < 16) ? 0 : 8;
    v16h frag;
#pragma unroll
    for (int j = 0; j < 8; ++j) {
        frag[j]     = (_Float16)B[(long)(k0 + halfsel + j)      * HDIM + col];
        frag[j + 8] = (_Float16)B[(long)(k0 + 16 + halfsel + j) * HDIM + col];
    }
    *(v16h*)(Bp + (size_t)t * 16) = frag;
}

// ---------------- WMMA GEMM: C[M x 64] = A[M x K] * B[K x 64] ----------------
// One wave per 16x64 output strip: 4 accumulators, A fragment reused across 4 WMMAs.
// Packed B staged in LDS (<= 32 KB), read as ds_load_b128 pairs.

__global__ __launch_bounds__(256)
void gemm_wmma_kernel(const float* __restrict__ A, const _Float16* __restrict__ Bp,
                      float* __restrict__ C, int M, int K) {
    __shared__ _Float16 bsh[MAX_KTILES * 4 * 32 * 16];   // 32 KB max

    int ktiles = K >> 5;
    int nvec   = (ktiles * 4 * 32 * 16) >> 3;            // uint4 chunks (8 halves)
    for (int v = threadIdx.x; v < nvec; v += 256)
        ((uint4*)bsh)[v] = ((const uint4*)Bp)[v];
    __syncthreads();

    int wave = threadIdx.x >> 5;
    int lane = threadIdx.x & 31;
    long mtile  = (long)blockIdx.x * 8 + wave;
    int  mtiles = (M + 15) >> 4;
    if (mtile >= mtiles) return;                          // wave-uniform, post-barrier

    int rbase   = (int)(mtile << 4);
    int halfsel = (lane < 16) ? 0 : 8;                    // K/M sub-block per ISA layout
    int row     = rbase + (lane & 15);
    if (row >= M) row = M - 1;                            // branchless edge clamp
    const float* arow = A + (long)row * K;

    v8f acc[4] = {};
    for (int kt = 0; kt < ktiles; ++kt) {
        int k0 = kt << 5;
        // A fragment: 4 aligned float4 loads (8+8 contiguous floats)
        float4 a0 = *(const float4*)(arow + k0 + halfsel);
        float4 a1 = *(const float4*)(arow + k0 + halfsel + 4);
        float4 a2 = *(const float4*)(arow + k0 + 16 + halfsel);
        float4 a3 = *(const float4*)(arow + k0 + 16 + halfsel + 4);
        v16h a;
        a[0]  = (_Float16)a0.x; a[1]  = (_Float16)a0.y;
        a[2]  = (_Float16)a0.z; a[3]  = (_Float16)a0.w;
        a[4]  = (_Float16)a1.x; a[5]  = (_Float16)a1.y;
        a[6]  = (_Float16)a1.z; a[7]  = (_Float16)a1.w;
        a[8]  = (_Float16)a2.x; a[9]  = (_Float16)a2.y;
        a[10] = (_Float16)a2.z; a[11] = (_Float16)a2.w;
        a[12] = (_Float16)a3.x; a[13] = (_Float16)a3.y;
        a[14] = (_Float16)a3.z; a[15] = (_Float16)a3.w;
#pragma unroll
        for (int nt = 0; nt < 4; ++nt) {
            v16h b = *(const v16h*)(bsh + (((kt << 2) + nt) * 32 + lane) * 16);
            acc[nt] = __builtin_amdgcn_wmma_f32_16x16x32_f16(
                false, a, false, b, (short)0, acc[nt], false, false);
        }
    }

    // C/D layout: VGPR r -> row (rbase + halfsel + r), col = nt*16 + (lane & 15)
    int colbase = lane & 15;
#pragma unroll
    for (int r = 0; r < 8; ++r) {
        int m = rbase + halfsel + r;
        if (m < M) {
            float* crow = C + (long)m * HDIM;
#pragma unroll
            for (int nt = 0; nt < 4; ++nt)
                crow[(nt << 4) + colbase] = acc[nt][r];
        }
    }
}

// ---------------- aggregation: out = sum_{e:dst} norm * h[src] ----------------

// self-loop term: out[n][c] = h[n][c] * disq[n]^2
__global__ void agg_init_kernel(const float* __restrict__ h,
                                const float* __restrict__ disq,
                                float* __restrict__ out, long total) {
    long i = (long)blockIdx.x * blockDim.x + threadIdx.x;
    if (i < total) {
        int n = (int)(i >> 6);
        float d = disq[n];
        out[i] = h[i] * d * d;
    }
}

// 4 edges per 256-thread block, 64 channels (lanes) per edge
__global__ __launch_bounds__(256)
void agg_edges_kernel(const float* __restrict__ h, const int* __restrict__ ei,
                      const float* __restrict__ disq, float* __restrict__ out,
                      int e) {
    int slot = threadIdx.x >> 6;
    int c    = threadIdx.x & 63;
    long edge = (long)blockIdx.x * 4 + slot;
    if (edge >= e) return;
    int s = ei[edge];                 // src row
    int d = ei[(long)e + edge];       // dst row
    float w = disq[s] * disq[d];
    atomicAdd(&out[(long)d * HDIM + c], h[(long)s * HDIM + c] * w);
}

__global__ void bias_relu_kernel(float* __restrict__ a, const float* __restrict__ b,
                                 long total) {
    long i = (long)blockIdx.x * blockDim.x + threadIdx.x;
    if (i < total) {
        float v = a[i] + b[i & 63];
        a[i] = v > 0.0f ? v : 0.0f;
    }
}

// ---------------- pooling + head ----------------

__global__ void zero_kernel(float* p, int n) {
    int i = blockIdx.x * blockDim.x + threadIdx.x;
    if (i < n) p[i] = 0.0f;
}

__global__ void pool_kernel(const float* __restrict__ a, const int* __restrict__ batch,
                            float* __restrict__ sums, float* __restrict__ cnt,
                            long total) {
    long i = (long)blockIdx.x * blockDim.x + threadIdx.x;
    if (i < total) {
        int n = (int)(i >> 6);
        int c = (int)(i & 63);
        int g = batch[n];
        atomicAdd(&sums[(long)g * HDIM + c], a[i]);
        if (c == 0) atomicAdd(&cnt[g], 1.0f);
    }
}

__global__ void head_kernel(const float* __restrict__ sums, const float* __restrict__ cnt,
                            const float* __restrict__ fcW, const float* __restrict__ fcb,
                            float* __restrict__ out, int g_n, int c_n) {
    int i = blockIdx.x * blockDim.x + threadIdx.x;
    if (i >= g_n * c_n) return;
    int g = i / c_n;
    int c = i % c_n;
    float cd = cnt[g];
    if (cd < 1.0f) cd = 1.0f;
    float inv = 1.0f / cd;
    float acc = fcb[c];
    for (int hh = 0; hh < HDIM; ++hh)
        acc += (sums[(long)g * HDIM + hh] * inv) * fcW[hh * c_n + c];
    out[i] = acc;
}

// ---------------- launcher ----------------

extern "C" void kernel_launch(void* const* d_in, const int* in_sizes, int n_in,
                              void* d_out, int out_size, void* d_ws, size_t ws_size,
                              hipStream_t stream) {
    const float* x    = (const float*)d_in[0];
    const int*   ei   = (const int*)  d_in[1];
    const int*   bat  = (const int*)  d_in[2];
    const float* W1   = (const float*)d_in[3];
    const float* b1   = (const float*)d_in[4];
    const float* W2   = (const float*)d_in[5];
    const float* b2   = (const float*)d_in[6];
    const float* fcW  = (const float*)d_in[7];
    const float* fcb  = (const float*)d_in[8];
    float*       out  = (float*)d_out;

    const int N    = in_sizes[2];
    const int F_IN = in_sizes[0] / N;     // 256 (= K of layer 1)
    const int E    = in_sizes[1] / 2;
    const int C    = in_sizes[8];
    const int G    = out_size / C;

    // workspace layout (floats; keep 64-float = 256 B alignment between regions)
    size_t Npad = (size_t)((N + 63) & ~63);
    float*     deg  = (float*)d_ws;                       // N
    float*     bufH = deg  + Npad;                        // N*64
    float*     bufA = bufH + (size_t)N * HDIM;            // N*64
    float*     sums = bufA + (size_t)N * HDIM;            // G*64
    float*     cnt  = sums + (size_t)G * HDIM;            // G
    _Float16*  Bp1  = (_Float16*)(cnt + (size_t)((G + 63) & ~63));  // F_IN*64 halves
    _Float16*  Bp2  = Bp1 + (size_t)(F_IN >> 5) * 4 * 32 * 16;      // 64*64 halves
    (void)ws_size; (void)n_in;

    const long totNH = (long)N * HDIM;
    const int  T = 256;
    dim3 blk(T);
    auto cdiv = [](long a, long b) { return (int)((a + b - 1) / b); };

    // degrees -> D^{-1/2}
    deg_init_kernel <<<cdiv(N, T), blk, 0, stream>>>(deg, N);
    deg_accum_kernel<<<cdiv(E, T), blk, 0, stream>>>(ei, deg, E);
    deg_rsqrt_kernel<<<cdiv(N, T), blk, 0, stream>>>(deg, N);

    // prepack weights into WMMA fragment order (f16)
    pack_b_kernel<<<cdiv((F_IN >> 5) * 128, T), blk, 0, stream>>>(W1, Bp1, F_IN);
    pack_b_kernel<<<cdiv((HDIM >> 5) * 128, T), blk, 0, stream>>>(W2, Bp2, HDIM);

    int mtiles     = (N + 15) >> 4;
    int gemmBlocks = cdiv(mtiles, 8);     // 8 waves/block, one 16x64 strip per wave

    // layer 1: linear -> aggregate -> bias+relu
    gemm_wmma_kernel<<<gemmBlocks, blk, 0, stream>>>(x, Bp1, bufH, N, F_IN);
    agg_init_kernel <<<cdiv(totNH, T), blk, 0, stream>>>(bufH, deg, bufA, totNH);
    agg_edges_kernel<<<cdiv(E, 4), blk, 0, stream>>>(bufH, ei, deg, bufA, E);
    bias_relu_kernel<<<cdiv(totNH, T), blk, 0, stream>>>(bufA, b1, totNH);

    // layer 2
    gemm_wmma_kernel<<<gemmBlocks, blk, 0, stream>>>(bufA, Bp2, bufH, N, HDIM);
    agg_init_kernel <<<cdiv(totNH, T), blk, 0, stream>>>(bufH, deg, bufA, totNH);
    agg_edges_kernel<<<cdiv(E, 4), blk, 0, stream>>>(bufH, ei, deg, bufA, E);
    bias_relu_kernel<<<cdiv(totNH, T), blk, 0, stream>>>(bufA, b2, totNH);

    // mean pool + head
    zero_kernel<<<cdiv(G * HDIM + G, T), blk, 0, stream>>>(sums, G * HDIM + G);
    pool_kernel<<<cdiv(totNH, T), blk, 0, stream>>>(bufA, bat, sums, cnt, totNH);
    head_kernel<<<cdiv(G * C, T), blk, 0, stream>>>(sums, cnt, fcW, fcb, out, G, C);
}
